// PartialCorr_37744172597313
// MI455X (gfx1250) — matrix operand order repaired and checked by reading, not compiled, not run on hardware
//
#include <hip/hip_runtime.h>
#include <hip/hip_bf16.h>

#define HW   3136   // 56*56
#define CIN  256
#define CR   64

typedef __attribute__((ext_vector_type(16))) __bf16 v16bf;
typedef __attribute__((ext_vector_type(8)))  float  v8f;

union BF16x16 {
  v16bf v;
  unsigned short u[16];
  uint4 q[2];
};

// Native bf16 convert (lowers to v_cvt_pk_bf16_f32 on gfx1250).
__device__ __forceinline__ unsigned short f2bf(float f) {
  __bf16 h = (__bf16)f;
  return __builtin_bit_cast(unsigned short, h);
}
__device__ __forceinline__ float bf2f(unsigned short h) {
  return (float)__builtin_bit_cast(__bf16, h);
}

// -------- Kernel 0: pre-convert w21 (64x256 f32) into WMMA A-operand layout (bf16) ----
// Slot s = (kb*4 + mb)*32 + lane holds the 16 bf16 values lane needs for that
// (k-block, m-block): u[0..7] = K run at k0+half*8, u[8..15] = same run +16.
__global__ __launch_bounds__(256) void prep_w21_kernel(
    const float* __restrict__ w21, unsigned short* __restrict__ wa)
{
  const int slot = blockIdx.x * 256 + threadIdx.x;   // 1024 slots
  if (slot >= 1024) return;
  const int lane = slot & 31;
  const int mk   = slot >> 5;       // kb*4 + mb
  const int kb   = mk >> 2;
  const int mb   = mk & 3;
  const int half = lane >> 4;
  const int row  = mb * 16 + (lane & 15);
  const float* wr = w21 + row * CIN + kb * 32 + half * 8;
  unsigned short* dst = wa + (size_t)slot * 16;
  #pragma unroll
  for (int j = 0; j < 8; ++j) {
    dst[j]     = f2bf(wr[j]);
    dst[8 + j] = f2bf(wr[16 + j]);
  }
}

// -------- Kernel 1: conv21 (256->64 1x1) + BN + ReLU, via WMMA bf16 --------
// One wave per tile of 16 positions; computes all 64 output channels.
__global__ __launch_bounds__(256) void conv21_wmma_kernel(
    const float* __restrict__ x,            // [nt=64][256][3136]
    const unsigned short* __restrict__ wa,  // preconverted A operands (bf16)
    const float* __restrict__ g21,          // [64]
    const float* __restrict__ b21,          // [64]
    unsigned short* __restrict__ y)         // bf16 [nt=64][64][3136]
{
  const int lane = threadIdx.x & 31;
  const int wv   = threadIdx.x >> 5;
  const int tile = blockIdx.x * 8 + wv;        // 12544 tiles total
  const int nt   = tile / 196;
  const int p0   = (tile - nt * 196) * 16;     // position offset inside hw plane

  const int col  = lane & 15;                  // N index (position within tile)
  const int half = lane >> 4;                  // lane half selects K range

  const float* xb = x + (size_t)nt * CIN * HW + p0 + col;
  const uint4* wa4 = (const uint4*)wa;         // 2 uint4 per slot

  v8f acc[4];
  #pragma unroll
  for (int m = 0; m < 4; ++m)
    #pragma unroll
    for (int j = 0; j < 8; ++j) acc[m][j] = 0.0f;

  for (int kb = 0; kb < 8; ++kb) {
    const int k0 = kb * 32;

    // B operand: 32x16 bf16. lane half h holds K=k0+8h..k0+8h+7 then K=k0+16+8h..k0+16+8h+7.
    // x is streamed exactly once -> non-temporal (preserve L2 for y / weight tables).
    BF16x16 bop;
    {
      const int kbase = k0 + half * 8;
      #pragma unroll
      for (int j = 0; j < 16; ++j)
        bop.v[j] = (__bf16)__builtin_nontemporal_load(&xb[(size_t)(kbase + (j < 8 ? j : j + 8)) * HW]);
    }

    // A operands: preconverted, two b128 loads each (L1/L2-hot 32KB table).
    #pragma unroll
    for (int mb = 0; mb < 4; ++mb) {
      const uint4* src = wa4 + (size_t)((kb * 4 + mb) * 32 + lane) * 2;
      BF16x16 aop;
      aop.q[0] = src[0];
      aop.q[1] = src[1];
      acc[mb] = __builtin_amdgcn_wmma_f32_16x16x32_bf16(
          false, aop.v, false, bop.v, (short)0, acc[mb], false, false);
    }
  }

  // BN + ReLU + store bf16. C layout: VGPR i -> M = i + 8*half, N = col.
  const float inv = rsqrtf(1.0f + 1e-5f);
  unsigned short* yb = y + (size_t)nt * CR * HW + p0 + col;
  #pragma unroll
  for (int mb = 0; mb < 4; ++mb) {
    #pragma unroll
    for (int i = 0; i < 8; ++i) {
      const int o = mb * 16 + i + half * 8;
      float v = acc[mb][i] * (g21[o] * inv) + b21[o];
      v = fmaxf(v, 0.0f);
      yb[(size_t)o * HW] = f2bf(v);   // regular store: y is re-read by kernel 2
    }
  }
}

// -------- Kernel 2: 3x3 correlation + BN + ReLU + conv22 (9->256) + BN + residual + ReLU ----
__global__ __launch_bounds__(256) void corr_conv22_kernel(
    const unsigned short* __restrict__ y,   // bf16 [64][64][3136]
    const float* __restrict__ x,            // residual [64][256][3136]
    const float* __restrict__ g22, const float* __restrict__ b22,   // [9]
    const float* __restrict__ w2,           // [256][9]
    const float* __restrict__ g23, const float* __restrict__ b23,   // [256]
    float* __restrict__ out)                // [64][256][3136]
{
  __shared__ float s1[16][16][16];   // [c][ty][tx]
  __shared__ float s2[16][18][19];   // [c][row][col] halo tile (time-shifted)
  __shared__ float sw2[256 * 9];
  __shared__ float sg23[256], sb23[256];

  const int tid = threadIdx.x;
  const int tx  = tid & 15, ty = tid >> 4;
  const int bt  = blockIdx.x;
  const int nt  = bt >> 4;
  const int tile = bt & 15;
  const int th0 = (tile >> 2) * 16;
  const int tw0 = (tile & 3) * 16;
  const int h = th0 + ty, wq = tw0 + tx;
  const bool valid = (h < 56) && (wq < 56);

  const float inv = rsqrtf(1.0f + 1e-5f);

  for (int i = tid; i < 2304; i += 256) sw2[i] = w2[i];
  sg23[tid] = g23[tid] * inv;
  sb23[tid] = b23[tid];

  const int t   = nt & 7;
  const int t2  = (t + 1 < 8) ? (t + 1) : 7;  // idx = min(arange(1,T+1), T-1)
  const int nt2 = nt - t + t2;

  const unsigned short* y1b = y + (size_t)nt  * CR * HW;
  const unsigned short* y2b = y + (size_t)nt2 * CR * HW;

  float cv[9];
  #pragma unroll
  for (int k = 0; k < 9; ++k) cv[k] = 0.0f;

  for (int cc = 0; cc < 4; ++cc) {
    const int c0 = cc * 16;
    __syncthreads();
    // center tile for this image/time
    #pragma unroll 4
    for (int c = 0; c < 16; ++c) {
      float v = 0.0f;
      if (valid) v = bf2f(y1b[(size_t)(c0 + c) * HW + h * 56 + wq]);
      s1[c][ty][tx] = v;
    }
    // halo tile from time-shifted image (zero padding outside 56x56)
    for (int idx = tid; idx < 16 * 18 * 18; idx += 256) {
      const int c   = idx / 324;
      const int rem = idx - c * 324;
      const int r   = rem / 18;
      const int q   = rem - r * 18;
      const int gh  = th0 + r - 1;
      const int gw  = tw0 + q - 1;
      float v = 0.0f;
      if ((unsigned)gh < 56u && (unsigned)gw < 56u)
        v = bf2f(y2b[(size_t)(c0 + c) * HW + gh * 56 + gw]);
      s2[c][r][q] = v;
    }
    __syncthreads();
    #pragma unroll 4
    for (int c = 0; c < 16; ++c) {
      const float a = s1[c][ty][tx];
      #pragma unroll
      for (int dy = 0; dy < 3; ++dy)
        #pragma unroll
        for (int dx = 0; dx < 3; ++dx)
          cv[dy * 3 + dx] += a * s2[c][ty + dy][tx + dx];
    }
  }
  __syncthreads();

  // BN22 + ReLU (divide by cr=64 first)
  #pragma unroll
  for (int k = 0; k < 9; ++k) {
    float v = (cv[k] * (1.0f / 64.0f)) * (g22[k] * inv) + b22[k];
    cv[k] = fmaxf(v, 0.0f);
  }

  if (valid) {
    const size_t pofs = (size_t)h * 56 + wq;
    const float* xr = x   + (size_t)nt * CIN * HW + pofs;
    float*       ob = out + (size_t)nt * CIN * HW + pofs;
    for (int o = 0; o < 256; ++o) {
      const float* wr = &sw2[o * 9];
      float a = 0.0f;
      #pragma unroll
      for (int k = 0; k < 9; ++k) a += wr[k] * cv[k];
      // residual x is read exactly once, out written exactly once -> non-temporal
      float v = a * sg23[o] + sb23[o] + __builtin_nontemporal_load(&xr[(size_t)o * HW]);
      __builtin_nontemporal_store(fmaxf(v, 0.0f), &ob[(size_t)o * HW]);
    }
  }
}

extern "C" void kernel_launch(void* const* d_in, const int* in_sizes, int n_in,
                              void* d_out, int out_size, void* d_ws, size_t ws_size,
                              hipStream_t stream) {
  (void)in_sizes; (void)n_in; (void)out_size; (void)ws_size;
  const float* x   = (const float*)d_in[0];
  const float* w21 = (const float*)d_in[1];
  const float* g21 = (const float*)d_in[2];
  const float* b21 = (const float*)d_in[3];
  const float* g22 = (const float*)d_in[4];
  const float* b22 = (const float*)d_in[5];
  const float* w22 = (const float*)d_in[6];
  const float* g23 = (const float*)d_in[7];
  const float* b23 = (const float*)d_in[8];
  float* out = (float*)d_out;

  // Workspace layout:
  //   [0, 25690112)           y (bf16)  : 64*64*3136*2 bytes
  //   [25690112, +32768)      wa (bf16) : preconverted conv21 weights, A-operand layout
  unsigned short* y  = (unsigned short*)d_ws;
  unsigned short* wa = (unsigned short*)((char*)d_ws + (size_t)64 * CR * HW * 2);

  prep_w21_kernel<<<4, 256, 0, stream>>>(w21, wa);
  // 64 images * 196 position-tiles = 12544 tiles; 8 waves/block -> 1568 blocks
  conv21_wmma_kernel<<<1568, 256, 0, stream>>>(x, wa, g21, b21, y);
  // 64 images * 16 spatial tiles = 1024 blocks
  corr_conv22_kernel<<<1024, 256, 0, stream>>>(y, x, g22, b22, w22, g23, b23, out);
}
